// DeformableTransformer_37254546325716
// MI455X (gfx1250) — matrix operand order repaired and checked by reading, not compiled, not run on hardware
//
#include <hip/hip_runtime.h>
#include <math.h>

// ---------------------------------------------------------------------------
// Deformable transformer decoder (6 layers) for MI455X / gfx1250.
// All GEMMs run on the WMMA pipe: v_wmma_f32_16x16x32_f16, fp32 accumulate.
// 128x64 block tile, 8 wave32s, each wave owns a 32x32 tile (4 WMMA / K-step).
// Double-buffered LDS staging: stage tile t+1 while WMMAs consume tile t.
// All boundary paths are branch-free (clamped loads + select-zero).
// ---------------------------------------------------------------------------

typedef __attribute__((ext_vector_type(16))) _Float16 v16h;
typedef __attribute__((ext_vector_type(8)))  _Float16 v8h;
typedef __attribute__((ext_vector_type(4)))  _Float16 v4h;
typedef __attribute__((ext_vector_type(8)))  float    v8f;
typedef __attribute__((ext_vector_type(4)))  float    v4f;

#define NLAYERS 6
#define BATCH   8
#define NQ      900
#define DIM     256
#define NHEAD   8
#define DHEAD   32
#define NLVL    4
#define NPTS    4
#define DFFN    1024
#define STOT    13294   // 100*100 + 50*50 + 25*25 + 13*13

#define BM 128
#define BN 64
#define BK 32
#define LDP 40          // padded LDS row stride in halves (80 B, 16B-aligned rows)

__device__ __forceinline__ v16h make16(v8h lo, v8h hi)
{
    return __builtin_shufflevector(lo, hi, 0, 1, 2, 3, 4, 5, 6, 7,
                                   8, 9, 10, 11, 12, 13, 14, 15);
}

// A fragment (16x32 f16): lane m = lane&15, kh = lane>>4.
__device__ __forceinline__ v16h load_afrag(const _Float16* row, int kh)
{
    v8h lo = *(const v8h*)(row + kh * 8);
    v8h hi = *(const v8h*)(row + 16 + kh * 8);
    return make16(lo, hi);
}

// B fragment (32x16 f16): lane n = lane&15, kh = lane>>4.
__device__ __forceinline__ v16h load_bfrag(const _Float16* row, int kh)
{
    v8h lo = *(const v8h*)(row + kh * 8);
    v8h hi = *(const v8h*)(row + 16 + kh * 8);
    return make16(lo, hi);
}

// ---------------------------------------------------------------------------
// Stage one (BM x BK) A tile + (BN x BK) B tile into LDS as f16.
// Fast path: float4 global loads + packed 8B LDS stores.
// Edge path: branch-free clamped loads + select-zero (loads pipeline freely).
// ---------------------------------------------------------------------------
__device__ __forceinline__ void stage_tile(
    _Float16 (*As)[LDP], _Float16 (*Bs)[LDP],
    const float* __restrict__ A, int lda,
    const float* __restrict__ B, int ldb, int transB,
    int m0, int n0, int k0, int M, int N, int K, int tid)
{
    // ---------------- A tile (128 x 32) ----------------
    if (m0 + BM <= M && k0 + BK <= K) {
        #pragma unroll
        for (int i = 0; i < 4; ++i) {
            int f = tid + i * 256;
            int r = f >> 3, c = (f & 7) * 4;
            v4f v = *(const v4f*)(A + (long)(m0 + r) * lda + (k0 + c));
            *(v4h*)&As[r][c] = __builtin_convertvector(v, v4h);
        }
    } else {
        #pragma unroll
        for (int i = 0; i < 16; ++i) {
            int f = tid + i * 256;
            int r = f >> 5, c = f & 31;
            int gm = m0 + r, gk = k0 + c;
            int gmc = gm < M ? gm : M - 1;
            int gkc = gk < K ? gk : K - 1;
            float v = A[(long)gmc * lda + gkc];
            As[r][c] = (gm < M && gk < K) ? (_Float16)v : (_Float16)0.0f;
        }
    }
    // ---------------- B tile (64 x 32) ----------------
    if (!transB) {
        if (n0 + BN <= N && k0 + BK <= K) {
            #pragma unroll
            for (int i = 0; i < 2; ++i) {
                int f = tid + i * 256;
                int r = f >> 3, c = (f & 7) * 4;
                v4f v = *(const v4f*)(B + (long)(n0 + r) * ldb + (k0 + c));
                *(v4h*)&Bs[r][c] = __builtin_convertvector(v, v4h);
            }
        } else {
            #pragma unroll
            for (int i = 0; i < 8; ++i) {
                int f = tid + i * 256;
                int r = f >> 5, c = f & 31;
                int gn = n0 + r, gk = k0 + c;
                int gnc = gn < N ? gn : N - 1;
                int gkc = gk < K ? gk : K - 1;
                float v = B[(long)gnc * ldb + gkc];
                Bs[r][c] = (gn < N && gk < K) ? (_Float16)v : (_Float16)0.0f;
            }
        }
    } else {
        // B stored [K,N]; scatter into Bs[n][k]; branch-free clamped loads
        #pragma unroll
        for (int i = 0; i < 2; ++i) {
            int f  = tid + i * 256;
            int k  = f >> 4, n4 = (f & 15) * 4;
            int gk = k0 + k;
            int gkc = gk < K ? gk : K - 1;
            #pragma unroll
            for (int j = 0; j < 4; ++j) {
                int gn  = n0 + n4 + j;
                int gnc = gn < N ? gn : N - 1;
                float v = B[(long)gkc * ldb + gnc];
                Bs[n4 + j][k] = (gk < K && gn < N) ? (_Float16)v
                                                   : (_Float16)0.0f;
            }
        }
    }
}

// ---------------------------------------------------------------------------
// Generic tiled GEMM:  C[M,N] = alpha * A[M,K] @ op(B)[K,N] (+bias) (+relu)
//   transB==0: B stored [N,K] row-major (computes A @ B^T)
//   transB==1: B stored [K,N] row-major (ldb = row stride)
// Batched via blockIdx.z with element strides.
// ---------------------------------------------------------------------------
__global__ __launch_bounds__(256) void gemm_wmma_kernel(
    const float* __restrict__ A, int lda, long strideA,
    const float* __restrict__ B, int ldb, long strideB, int transB,
    const float* __restrict__ bias,
    float* __restrict__ C, int ldc, long strideC,
    int M, int N, int K, float alpha, int relu)
{
    __shared__ _Float16 As[2][BM][LDP];
    __shared__ _Float16 Bs[2][BN][LDP];

    const int batch = blockIdx.z;
    A += (long)batch * strideA;
    B += (long)batch * strideB;
    C += (long)batch * strideC;

    const int m0   = blockIdx.y * BM;
    const int n0   = blockIdx.x * BN;
    const int tid  = threadIdx.x;
    const int lane = tid & 31;
    const int wave = tid >> 5;         // 0..7
    const int wrow = wave >> 1;        // 0..3 : 32-row strip
    const int wcol = wave & 1;         // 0..1 : 32-col strip

    v8f acc00 = {}, acc01 = {}, acc10 = {}, acc11 = {};

    const int mrow = lane & 15;
    const int kh   = lane >> 4;

    const int nk = (K + BK - 1) / BK;

    stage_tile(As[0], Bs[0], A, lda, B, ldb, transB,
               m0, n0, 0, M, N, K, tid);
    __syncthreads();

    for (int t = 0; t < nk; ++t) {
        const int cur = t & 1;
        if (t + 1 < nk)
            stage_tile(As[cur ^ 1], Bs[cur ^ 1], A, lda, B, ldb, transB,
                       m0, n0, (t + 1) * BK, M, N, K, tid);

        v16h a0 = load_afrag(&As[cur][wrow * 32 + mrow][0], kh);
        v16h a1 = load_afrag(&As[cur][wrow * 32 + 16 + mrow][0], kh);
        v16h b0 = load_bfrag(&Bs[cur][wcol * 32 + mrow][0], kh);
        v16h b1 = load_bfrag(&Bs[cur][wcol * 32 + 16 + mrow][0], kh);

        acc00 = __builtin_amdgcn_wmma_f32_16x16x32_f16(
                    false, a0, false, b0, (short)0, acc00, false, false);
        acc01 = __builtin_amdgcn_wmma_f32_16x16x32_f16(
                    false, a0, false, b1, (short)0, acc01, false, false);
        acc10 = __builtin_amdgcn_wmma_f32_16x16x32_f16(
                    false, a1, false, b0, (short)0, acc10, false, false);
        acc11 = __builtin_amdgcn_wmma_f32_16x16x32_f16(
                    false, a1, false, b1, (short)0, acc11, false, false);
        __syncthreads();
    }

    // ---- epilogue: lane holds col n = lane&15, VGPR r -> M = r + 8*(lane>>4)
    const int ncol = lane & 15;
    const int rhal = (lane >> 4) * 8;
    #pragma unroll
    for (int i = 0; i < 2; ++i) {
        #pragma unroll
        for (int r = 0; r < 8; ++r) {
            int gm = m0 + wrow * 32 + i * 16 + rhal + r;
            if (gm >= M) continue;
            #pragma unroll
            for (int j = 0; j < 2; ++j) {
                int gn = n0 + wcol * 32 + j * 16 + ncol;
                if (gn >= N) continue;
                v8f* accp = (i == 0) ? (j == 0 ? &acc00 : &acc01)
                                     : (j == 0 ? &acc10 : &acc11);
                float v = (*accp)[r] * alpha;
                if (bias) v += bias[gn];
                if (relu) v = v > 0.0f ? v : 0.0f;
                C[(long)gm * ldc + gn] = v;
            }
        }
    }
}

// ---------------------------------------------------------------------------
// Elementwise add: o = a + b
// ---------------------------------------------------------------------------
__global__ void add_kernel(const float* __restrict__ a,
                           const float* __restrict__ b,
                           float* __restrict__ o, long n)
{
    long i = (long)blockIdx.x * blockDim.x + threadIdx.x;
    if (i < n) o[i] = a[i] + b[i];
}

// ---------------------------------------------------------------------------
// inter_ref: replicate reference_points NL times
// ---------------------------------------------------------------------------
__global__ void fill_ref_kernel(const float* __restrict__ ref,
                                float* __restrict__ out, long per, long total)
{
    long i = (long)blockIdx.x * blockDim.x + threadIdx.x;
    if (i < total) out[i] = ref[i % per];
}

// ---------------------------------------------------------------------------
// Row softmax (row length = rowlen), one 256-thread block per row
// ---------------------------------------------------------------------------
__global__ __launch_bounds__(256) void softmax_rows_kernel(float* __restrict__ x,
                                                           int rowlen)
{
    long row = blockIdx.x;
    float* p = x + row * (long)rowlen;
    __shared__ float red[256];
    int tid = threadIdx.x;

    float m = -1e30f;
    for (int i = tid; i < rowlen; i += 256) m = fmaxf(m, p[i]);
    red[tid] = m; __syncthreads();
    for (int s = 128; s > 0; s >>= 1) {
        if (tid < s) red[tid] = fmaxf(red[tid], red[tid + s]);
        __syncthreads();
    }
    m = red[0]; __syncthreads();

    float sum = 0.0f;
    for (int i = tid; i < rowlen; i += 256) {
        float e = __expf(p[i] - m);
        p[i] = e;
        sum += e;
    }
    red[tid] = sum; __syncthreads();
    for (int s = 128; s > 0; s >>= 1) {
        if (tid < s) red[tid] += red[tid + s];
        __syncthreads();
    }
    float inv = 1.0f / red[0];
    for (int i = tid; i < rowlen; i += 256) p[i] *= inv;
}

// ---------------------------------------------------------------------------
// Sampling-weight softmax: per (b,q), 8 heads x 16 weights, softmax over 16
// ---------------------------------------------------------------------------
__global__ __launch_bounds__(128) void aw_softmax_kernel(float* __restrict__ aw)
{
    long row = blockIdx.x;                 // b*Q + q
    float* p = aw + row * 128;
    __shared__ float s[128];
    int tid = threadIdx.x;
    s[tid] = p[tid];
    __syncthreads();
    int g = (tid >> 4) * 16;
    float m = -1e30f;
    for (int i = 0; i < 16; ++i) m = fmaxf(m, s[g + i]);
    float sum = 0.0f;
    for (int i = 0; i < 16; ++i) sum += __expf(s[g + i] - m);
    p[tid] = __expf(s[tid] - m) / sum;
}

// ---------------------------------------------------------------------------
// LayerNorm of (a + r) over last dim (256), one block per row
// ---------------------------------------------------------------------------
__global__ __launch_bounds__(256) void layernorm_kernel(
    const float* __restrict__ a, const float* __restrict__ r,
    const float* __restrict__ g, const float* __restrict__ bta,
    float* __restrict__ out)
{
    long row = blockIdx.x;
    int tid = threadIdx.x;
    __shared__ float red[256];
    float x = a[row * 256 + tid] + r[row * 256 + tid];
    red[tid] = x; __syncthreads();
    for (int s = 128; s > 0; s >>= 1) {
        if (tid < s) red[tid] += red[tid + s];
        __syncthreads();
    }
    float mean = red[0] * (1.0f / 256.0f);
    __syncthreads();
    float d = x - mean;
    red[tid] = d * d; __syncthreads();
    for (int s = 128; s > 0; s >>= 1) {
        if (tid < s) red[tid] += red[tid + s];
        __syncthreads();
    }
    float var = red[0] * (1.0f / 256.0f);
    out[row * 256 + tid] = d * rsqrtf(var + 1e-5f) * g[tid] + bta[tid];
}

// ---------------------------------------------------------------------------
// Deformable bilinear sampling + weighted sum.
// grid: B*Q blocks; 256 threads = (head h = tid>>5, channel d = tid&31)
// Also emits locs[b,lid,q,h,l,p,2] and aws[b,lid,q,h,l,p].
// ---------------------------------------------------------------------------
__device__ __forceinline__ float dsamp(const float* __restrict__ v, int b,
                                       int st, int H, int W, int xi, int yi,
                                       int c)
{
    // branch-free: clamp (always-legal load) + select-zero, matches the
    // reference's clip + valid-mask semantics exactly
    int xic = xi < 0 ? 0 : (xi >= W ? W - 1 : xi);
    int yic = yi < 0 ? 0 : (yi >= H ? H - 1 : yi);
    long idx = (long)b * STOT + st + (long)yic * W + xic;
    float val = v[idx * DIM + c];
    bool ok = (xi >= 0) & (xi < W) & (yi >= 0) & (yi < H);
    return ok ? val : 0.0f;
}

__global__ __launch_bounds__(256) void deform_sample_kernel(
    const float* __restrict__ ref,     // [B,Q,2]
    const float* __restrict__ offb,    // [B,Q,256]
    const float* __restrict__ awb,     // [B,Q,128] (softmaxed)
    const float* __restrict__ vbuf,    // [B,S,256]
    float* __restrict__ ca,            // [B,Q,256]
    float* __restrict__ locs_out,      // full locs tensor base
    float* __restrict__ aws_out,       // full aws tensor base
    int lid)
{
    const int Hs[4] = {100, 50, 25, 13};
    const int Ws[4] = {100, 50, 25, 13};
    const int St[4] = {0, 10000, 12500, 13125};

    long bq = blockIdx.x;
    int b = (int)(bq / NQ);
    int q = (int)(bq % NQ);
    int tid = threadIdx.x;
    int h = tid >> 5;
    int d = tid & 31;
    int c = h * DHEAD + d;

    float rx = ref[bq * 2 + 0];
    float ry = ref[bq * 2 + 1];

    float acc = 0.0f;
    #pragma unroll
    for (int l = 0; l < NLVL; ++l) {
        const int Hh = Hs[l], Ww = Ws[l], st = St[l];
        #pragma unroll
        for (int p = 0; p < NPTS; ++p) {
            int oi = (h * NLVL + l) * NPTS + p;
            float ox = offb[bq * 256 + oi * 2 + 0];
            float oy = offb[bq * 256 + oi * 2 + 1];
            float lx = rx + ox / (float)Ww;
            float ly = ry + oy / (float)Hh;
            float a  = awb[bq * 128 + oi];

            if (d == 0) {
                long lo = ((((long)(b * NLAYERS + lid) * NQ + q) * NHEAD + h)
                           * NLVL + l) * NPTS + p;
                locs_out[lo * 2 + 0] = lx;
                locs_out[lo * 2 + 1] = ly;
                aws_out[lo] = a;
            }

            float x = lx * (float)Ww - 0.5f;
            float y = ly * (float)Hh - 0.5f;
            float x0 = floorf(x), y0 = floorf(y);
            float wx = x - x0, wy = y - y0;
            int x0i = (int)x0, y0i = (int)y0;

            float s00 = dsamp(vbuf, b, st, Hh, Ww, x0i,     y0i,     c);
            float s10 = dsamp(vbuf, b, st, Hh, Ww, x0i + 1, y0i,     c);
            float s01 = dsamp(vbuf, b, st, Hh, Ww, x0i,     y0i + 1, c);
            float s11 = dsamp(vbuf, b, st, Hh, Ww, x0i + 1, y0i + 1, c);

            float val = s00 * (1.0f - wx) * (1.0f - wy)
                      + s10 * wx * (1.0f - wy)
                      + s01 * (1.0f - wx) * wy
                      + s11 * wx * wy;
            acc += a * val;
        }
    }
    ca[bq * 256 + c] = acc;
}

// ---------------------------------------------------------------------------
// Host side
// ---------------------------------------------------------------------------
static void launch_gemm(const float* A, int lda, long sA,
                        const float* B, int ldb, long sB, int transB,
                        const float* bias, float* C, int ldc, long sC,
                        int M, int N, int K, float alpha, int relu,
                        int batch, hipStream_t stream)
{
    dim3 grid((N + BN - 1) / BN, (M + BM - 1) / BM, batch);
    gemm_wmma_kernel<<<grid, 256, 0, stream>>>(A, lda, sA, B, ldb, sB, transB,
                                               bias, C, ldc, sC, M, N, K,
                                               alpha, relu);
}

extern "C" void kernel_launch(void* const* d_in, const int* in_sizes, int n_in,
                              void* d_out, int out_size, void* d_ws, size_t ws_size,
                              hipStream_t stream)
{
    (void)in_sizes; (void)n_in; (void)out_size; (void)ws_size;

    const float* tgt      = (const float*)d_in[0];
    const float* refp     = (const float*)d_in[1];
    const float* src      = (const float*)d_in[2];
    const float* qpos     = (const float*)d_in[3];
    const float* sa_in_w  = (const float*)d_in[4];   // [6,768,256]
    const float* sa_in_b  = (const float*)d_in[5];   // [6,768]
    const float* sa_out_w = (const float*)d_in[6];   // [6,256,256]
    const float* sa_out_b = (const float*)d_in[7];
    const float* ln1_g    = (const float*)d_in[8];
    const float* ln1_b    = (const float*)d_in[9];
    const float* ln2_g    = (const float*)d_in[10];
    const float* ln2_b    = (const float*)d_in[11];
    const float* ln3_g    = (const float*)d_in[12];
    const float* ln3_b    = (const float*)d_in[13];
    const float* val_w    = (const float*)d_in[14];  // [6,256,256]
    const float* val_b    = (const float*)d_in[15];
    const float* off_w    = (const float*)d_in[16];  // [6,256,256]
    const float* off_b    = (const float*)d_in[17];
    const float* aw_w     = (const float*)d_in[18];  // [6,128,256]
    const float* aw_b     = (const float*)d_in[19];
    const float* cao_w    = (const float*)d_in[20];  // [6,256,256]
    const float* cao_b    = (const float*)d_in[21];
    const float* ffn_w1   = (const float*)d_in[22];  // [6,1024,256]
    const float* ffn_b1   = (const float*)d_in[23];
    const float* ffn_w2   = (const float*)d_in[24];  // [6,256,1024]
    const float* ffn_b2   = (const float*)d_in[25];

    const long BQ = (long)BATCH * NQ;                // 7200
    const long nBQD = BQ * DIM;                      // 1,843,200

    // ---- d_out partitions (float32, return-order concat) ----
    float* out      = (float*)d_out;
    float* inter    = out;                                   // [6,8,900,256]
    float* interref = inter + (long)NLAYERS * nBQD;          // [6,8,900,2]
    float* locs     = interref + (long)NLAYERS * BQ * 2;     // [8,6,900,8,4,4,2]
    float* aws      = locs + (long)BATCH * NLAYERS * NQ * NHEAD * NLVL * NPTS * 2;

    // ---- scratch partitions (floats) ----
    float* ws = (float*)d_ws;
    size_t o = 0;
    float* vbuf    = ws + o; o += (size_t)BATCH * STOT * DIM;  // value proj
    float* qkin    = ws + o; o += nBQD;                        // x + qpos
    float* qkbuf   = ws + o; o += BQ * 512;                    // q|k heads
    float* vhbuf   = ws + o; o += nBQD;                        // v heads
    float* scores  = ws + o; o += (size_t)NHEAD * NQ * NQ;     // per-batch scores
    float* attnout = ws + o; o += nBQD;
    float* tmpb    = ws + o; o += nBQD;                        // after ln2
    float* xbuf    = ws + o; o += nBQD;                        // after ln1
    float* offbuf  = ws + o; o += nBQD;
    float* awbuf   = ws + o; o += BQ * 128;
    float* capre   = ws + o; o += nBQD;
    float* ffnbuf  = ws + o; o += BQ * DFFN;
    float* sabuf   = ws + o; o += nBQD;

    // inter_ref (reference points never change)
    {
        long total = (long)NLAYERS * BQ * 2;
        fill_ref_kernel<<<dim3((unsigned)((total + 255) / 256)), 256, 0, stream>>>(
            refp, interref, BQ * 2, total);
    }

    const float scale = 0.17677669529663687f;   // 1/sqrt(32)
    const float* cur = tgt;

    for (int lid = 0; lid < NLAYERS; ++lid) {
        const float* Wi = sa_in_w + (long)lid * 768 * 256;
        const float* bi = sa_in_b + (long)lid * 768;

        // ---------- self attention ----------
        add_kernel<<<dim3((unsigned)((nBQD + 255) / 256)), 256, 0, stream>>>(
            cur, qpos, qkin, nBQD);

        // q|k projection: [7200,256] x Wi[0:512]^T -> [7200,512]
        launch_gemm(qkin, 256, 0, Wi, 256, 0, 0, bi,
                    qkbuf, 512, 0, (int)BQ, 512, 256, 1.0f, 0, 1, stream);
        // v projection: [7200,256] x Wi[512:768]^T -> [7200,256]
        launch_gemm(cur, 256, 0, Wi + 512L * 256, 256, 0, 0, bi + 512,
                    vhbuf, 256, 0, (int)BQ, 256, 256, 1.0f, 0, 1, stream);

        for (int b = 0; b < BATCH; ++b) {
            const float* qb = qkbuf + (long)b * NQ * 512;
            const float* kb = qb + 256;
            // scores[h] = (Q_h @ K_h^T) / sqrt(dh) : per-head strided batch
            launch_gemm(qb, 512, 32, kb, 512, 32, 0, nullptr,
                        scores, NQ, (long)NQ * NQ,
                        NQ, NQ, DHEAD, scale, 0, NHEAD, stream);
            softmax_rows_kernel<<<NHEAD * NQ, 256, 0, stream>>>(scores, NQ);
            // attn @ V  (V accessed transposed: transB=1)
            launch_gemm(scores, NQ, (long)NQ * NQ,
                        vhbuf + (long)b * NQ * DIM, 256, 32, 1, nullptr,
                        attnout + (long)b * NQ * DIM, 256, 32,
                        NQ, DHEAD, NQ, 1.0f, 0, NHEAD, stream);
        }
        // output projection
        launch_gemm(attnout, 256, 0, sa_out_w + (long)lid * 256 * 256, 256, 0, 0,
                    sa_out_b + (long)lid * 256, sabuf, 256, 0,
                    (int)BQ, 256, 256, 1.0f, 0, 1, stream);
        // norm2
        layernorm_kernel<<<(unsigned)BQ, 256, 0, stream>>>(
            cur, sabuf, ln2_g + (long)lid * 256, ln2_b + (long)lid * 256, tmpb);

        // ---------- deformable cross attention ----------
        add_kernel<<<dim3((unsigned)((nBQD + 255) / 256)), 256, 0, stream>>>(
            tmpb, qpos, qkin, nBQD);

        // value projection: [B*S,256] x val_w^T + val_b
        launch_gemm(src, 256, 0, val_w + (long)lid * 256 * 256, 256, 0, 0,
                    val_b + (long)lid * 256, vbuf, 256, 0,
                    BATCH * STOT, 256, 256, 1.0f, 0, 1, stream);
        // offsets
        launch_gemm(qkin, 256, 0, off_w + (long)lid * 256 * 256, 256, 0, 0,
                    off_b + (long)lid * 256, offbuf, 256, 0,
                    (int)BQ, 256, 256, 1.0f, 0, 1, stream);
        // attention weights
        launch_gemm(qkin, 256, 0, aw_w + (long)lid * 128 * 256, 256, 0, 0,
                    aw_b + (long)lid * 128, awbuf, 128, 0,
                    (int)BQ, 128, 256, 1.0f, 0, 1, stream);
        aw_softmax_kernel<<<(unsigned)BQ, 128, 0, stream>>>(awbuf);

        deform_sample_kernel<<<(unsigned)BQ, 256, 0, stream>>>(
            refp, offbuf, awbuf, vbuf, capre, locs, aws, lid);

        // cross-attn output projection
        launch_gemm(capre, 256, 0, cao_w + (long)lid * 256 * 256, 256, 0, 0,
                    cao_b + (long)lid * 256, attnout, 256, 0,
                    (int)BQ, 256, 256, 1.0f, 0, 1, stream);
        // norm1
        layernorm_kernel<<<(unsigned)BQ, 256, 0, stream>>>(
            tmpb, attnout, ln1_g + (long)lid * 256, ln1_b + (long)lid * 256, xbuf);

        // ---------- FFN ----------
        launch_gemm(xbuf, 256, 0, ffn_w1 + (long)lid * DFFN * 256, 256, 0, 0,
                    ffn_b1 + (long)lid * DFFN, ffnbuf, DFFN, 0,
                    (int)BQ, DFFN, 256, 1.0f, 1, 1, stream);
        launch_gemm(ffnbuf, DFFN, 0, ffn_w2 + (long)lid * 256 * DFFN, DFFN, 0, 0,
                    ffn_b2 + (long)lid * 256, sabuf, 256, 0,
                    (int)BQ, 256, DFFN, 1.0f, 0, 1, stream);

        // norm3 -> write straight into the inter[lid] output slice
        float* interSlice = inter + (long)lid * nBQD;
        layernorm_kernel<<<(unsigned)BQ, 256, 0, stream>>>(
            xbuf, sabuf, ln3_g + (long)lid * 256, ln3_b + (long)lid * 256,
            interSlice);
        cur = interSlice;
    }
}
